// attention_53652731462151
// MI455X (gfx1250) — matrix operand order repaired and checked by reading, not compiled, not run on hardware
//
#include <hip/hip_runtime.h>
#include <hip/hip_bf16.h>
#include <math.h>
#include <stdint.h>

// Problem constants (from reference)
static constexpr int Bc  = 16;    // batch
static constexpr int Sc  = 1194;  // sequence
static constexpr int Spc = 1216;  // padded sequence (38 * 32)
static constexpr int Dc  = 512;   // model dim
static constexpr int Hc  = 2;     // heads
static constexpr int HDc = 256;   // head dim

typedef __bf16 bf16_t;
typedef __attribute__((ext_vector_type(16))) __bf16 v16bf;
typedef __attribute__((ext_vector_type(8)))  __bf16 v8bf;
typedef __attribute__((ext_vector_type(8)))  float  v8f;
typedef unsigned int u32x4 __attribute__((ext_vector_type(4)));
typedef int          i32x4 __attribute__((ext_vector_type(4)));
typedef int          i32x8 __attribute__((ext_vector_type(8)));

static __device__ inline v8f vzero8() {
  v8f z = {0.f, 0.f, 0.f, 0.f, 0.f, 0.f, 0.f, 0.f};
  return z;
}

static __device__ inline v8f wmma_bf16(v16bf a, v16bf b, v8f c) {
  // (neg_a, A, neg_b, B, c_mod, C, reuse_a, reuse_b)
  return __builtin_amdgcn_wmma_f32_16x16x32_bf16(false, a, false, b, (short)0, c,
                                                 false, false);
}

// A fragment (16x32 bf16) from row-major source, row stride `ld` elements.
// ISA layout: lane L -> row = L%16; cols {8*(L/16)..+7} and {16+8*(L/16)..+7}.
static __device__ inline v16bf load_a16(const bf16_t* __restrict__ base, int ld, int lane) {
  int row = lane & 15;
  int c0  = (lane >> 4) << 3; // 0 or 8
  const bf16_t* p = base + row * ld + c0;
  union { v16bf v; v8bf h[2]; } u;
  u.h[0] = *(const v8bf*)(p);
  u.h[1] = *(const v8bf*)(p + 16);
  return u.v;
}

// B fragment (32x16 bf16) where element B[k][n] = base[n*ld + k].
// ISA layout: lane L -> col n = L%16; rows k {8*(L/16)..+7} and {16+8*(L/16)..+7}.
static __device__ inline v16bf load_b16(const bf16_t* __restrict__ base, int ld, int lane) {
  int n  = lane & 15;
  int k0 = (lane >> 4) << 3; // 0 or 8
  const bf16_t* p = base + n * ld + k0;
  union { v16bf v; v8bf h[2]; } u;
  u.h[0] = *(const v8bf*)(p);
  u.h[1] = *(const v8bf*)(p + 16);
  return u.v;
}

// ---------------------------------------------------------------------------
// Tensor Data Mover: 2-D tile load Global -> LDS (D# per CDNA5 ISA ch.8).
//   addr(x,y) = global_addr + 2*(x + y*stride0); LDS packed row-major with
//   optional padding of (pad_amount+1) dwords every 2^(pad_interval+1) dwords.
// ---------------------------------------------------------------------------
#define HAVE_TDM __has_builtin(__builtin_amdgcn_tensor_load_to_lds)

static __device__ inline unsigned lds_off(const void* p) {
  return (unsigned)(uintptr_t)p; // flat LDS addr: low 32 bits = group offset
}

static __device__ inline void tdm_load_2d(const bf16_t* gptr, unsigned lds_addr,
                                          unsigned tensor_d0, unsigned tensor_d1,
                                          unsigned tile_d0, unsigned tile_d1,
                                          unsigned stride0,
                                          unsigned pad_i, unsigned pad_a,
                                          bf16_t* lds_fallback, int lds_ld) {
#if HAVE_TDM
  (void)lds_fallback; (void)lds_ld;
  unsigned long long ga = (unsigned long long)(uintptr_t)gptr;
  u32x4 g0;
  g0[0] = 1u;                                             // count=1, user mode
  g0[1] = lds_addr;                                       // bits 63:32
  g0[2] = (unsigned)(ga & 0xFFFFFFFFu);                   // global_addr lo
  g0[3] = (unsigned)((ga >> 32) & 0x1FFFFFFu) | (2u << 30); // addr hi | type=2
  i32x8 g1;
  unsigned w0 = (1u << 16);                               // data_size=1 (2B)
  w0 |= (1u << 20) | (pad_i << 22) | (pad_a << 25);       // pad enable/cfg
  g1[0] = (int)w0;
  g1[1] = (int)((tensor_d0 & 0xFFFFu) << 16);             // dim0[15:0] @63:48
  g1[2] = (int)((tensor_d0 >> 16) | ((tensor_d1 & 0xFFFFu) << 16));
  g1[3] = (int)((tensor_d1 >> 16) | (tile_d0 << 16));     // tile_dim0 @127:112
  g1[4] = (int)tile_d1;                                   // tile_dim1 (tile_dim2=0)
  g1[5] = (int)stride0;                                   // dim0_stride lo
  g1[6] = 0;
  g1[7] = 0;
  i32x4 z4 = {0, 0, 0, 0};
#if defined(__clang_major__) && __clang_major__ >= 23
  i32x8 z8 = {0, 0, 0, 0, 0, 0, 0, 0};
  __builtin_amdgcn_tensor_load_to_lds(g0, g1, z4, z4, z8, 0);
#else
  __builtin_amdgcn_tensor_load_to_lds(g0, g1, z4, z4, 0);
#endif
#else
  // Cooperative fallback: issuing wave copies the tile (padded LDS rows).
  (void)lds_addr; (void)tensor_d0; (void)tensor_d1; (void)pad_i; (void)pad_a;
  int lane = threadIdx.x & 31;
  for (unsigned y = 0; y < tile_d1; ++y)
    for (unsigned x = lane; x < tile_d0; x += 32)
      lds_fallback[y * lds_ld + x] = gptr[(size_t)y * stride0 + x];
#endif
}

template <int N>
static __device__ inline void tdm_wait() {
#if HAVE_TDM
  __builtin_amdgcn_s_wait_tensorcnt(N); // immediate must be a literal constant
#endif
}

// ---------------------------------------------------------------------------
// Conversion kernels
// ---------------------------------------------------------------------------
__global__ void cvt_f32_to_bf16(const float* __restrict__ in, bf16_t* __restrict__ out, int n) {
  int i = blockIdx.x * 256 + threadIdx.x;
  if (i < n) out[i] = (bf16_t)in[i];
}

// in [rows, cols] fp32 -> out [cols, rows] bf16
__global__ void transpose_cvt(const float* __restrict__ in, bf16_t* __restrict__ out,
                              int rows, int cols) {
  int i = blockIdx.x * 256 + threadIdx.x;
  if (i >= rows * cols) return;
  int c = i / rows;
  int r = i - c * rows;
  out[(size_t)c * rows + r] = (bf16_t)in[(size_t)r * cols + c];
}

// ---------------------------------------------------------------------------
// QKV projection: qkv = X @ W_qkv + b ; scatter into Q (pre-scaled), K, V^T
// 48x64 output tile per wave: grid (19104/48 = 398, 1536/64 = 24), block 32.
// __launch_bounds__(32,1): one wave per block -> full VGPR budget, no spills.
// ---------------------------------------------------------------------------
__global__ void __launch_bounds__(32, 1)
qkv_gemm(const bf16_t* __restrict__ X,   // [B*S, 512] bf16
         const bf16_t* __restrict__ WT,  // [1536, 512] bf16 (W^T)
         const float*  __restrict__ bias,
         bf16_t* __restrict__ Q,         // [B,H,Sp,HD]
         bf16_t* __restrict__ K,         // [B,H,Sp,HD]
         bf16_t* __restrict__ VT) {      // [B,H,HD,Sp]
  int rt = blockIdx.x, ct = blockIdx.y;
  int lane = threadIdx.x & 31;
  const bf16_t* Xt = X  + (size_t)rt * 48 * Dc;
  const bf16_t* Wt = WT + (size_t)ct * 64 * Dc;
  v8f acc[3][4];
  for (int i = 0; i < 3; ++i)
    for (int j = 0; j < 4; ++j) acc[i][j] = vzero8();
  for (int kc = 0; kc < Dc / 32; ++kc) {
    v16bf a0 = load_a16(Xt + 0 * 16 * Dc + kc * 32, Dc, lane);
    v16bf a1 = load_a16(Xt + 1 * 16 * Dc + kc * 32, Dc, lane);
    v16bf a2 = load_a16(Xt + 2 * 16 * Dc + kc * 32, Dc, lane);
    for (int j = 0; j < 4; ++j) {
      v16bf b = load_b16(Wt + (size_t)j * 16 * Dc + kc * 32, Dc, lane);
      acc[0][j] = wmma_bf16(a0, b, acc[0][j]);
      acc[1][j] = wmma_bf16(a1, b, acc[1][j]);
      acc[2][j] = wmma_bf16(a2, b, acc[2][j]);
    }
  }
  int n = lane & 15, half = lane >> 4;
  const float qscale = 0.0625f; // 1/sqrt(256) folded into Q
  for (int i = 0; i < 3; ++i) {
    for (int j = 0; j < 4; ++j) {
      int c = ct * 64 + j * 16 + n;   // 0..1535
      float bv = bias[c];
      for (int r = 0; r < 8; ++r) {
        int g = rt * 48 + i * 16 + r + 8 * half;  // global row in [0, B*S)
        int bb = g / Sc, s = g - bb * Sc;
        float val = acc[i][j][r] + bv;
        if (c < Dc) {
          int hh = c >> 8, d = c & 255;
          Q[(((bb * Hc) + hh) * Spc + s) * HDc + d] = (bf16_t)(val * qscale);
        } else if (c < 2 * Dc) {
          int cc = c - Dc; int hh = cc >> 8, d = cc & 255;
          K[(((bb * Hc) + hh) * Spc + s) * HDc + d] = (bf16_t)val;
        } else {
          int cc = c - 2 * Dc; int hh = cc >> 8, d = cc & 255;
          VT[(((bb * Hc) + hh) * HDc + d) * Spc + s] = (bf16_t)val;
        }
      }
    }
  }
}

// ---------------------------------------------------------------------------
// Flash attention (causal, online softmax), K/V staged into LDS by the TDM.
// grid: (75, B*H = 32), block 64 (2 waves). Wave w accumulates O dims
// [w*128,(w+1)*128). K double-buffered (prefetch), V fetched under QK shadow.
// ---------------------------------------------------------------------------
__global__ void __launch_bounds__(64, 1)
flash_attn(const bf16_t* __restrict__ Q,
           const bf16_t* __restrict__ K,
           const bf16_t* __restrict__ VT,
           bf16_t* __restrict__ O) { // [B*S, 512]
  constexpr int KLD = 264; // 256 + 8 bf16 pad (16B per 512B row) -> conflict-free
  constexpr int VLD = 40;  // 32 + 8 bf16 pad (16B per 64B row)
  __shared__ __align__(16) bf16_t kbuf[2][32 * KLD];  // 2 x 16.5 KB
  __shared__ __align__(16) bf16_t vbuf[256 * VLD];    // 20 KB
  __shared__ __align__(16) bf16_t plds[2][16 * 40];   // P tiles

  int qt = blockIdx.x, bh = blockIdx.y;
  int lane = threadIdx.x & 31;
  int wave = threadIdx.x >> 5;
  int qbase = qt * 16;

  const bf16_t* Qp = Q  + ((size_t)bh * Spc + qbase) * HDc;
  const bf16_t* Kp = K  + (size_t)bh * Spc * HDc;
  const bf16_t* Vp = VT + (size_t)bh * HDc * Spc;

  v8f acc[8];
  for (int dt = 0; dt < 8; ++dt) acc[dt] = vzero8();
  float mi[8], li[8];
  for (int r = 0; r < 8; ++r) { mi[r] = -INFINITY; li[r] = 0.f; }

  int n = lane & 15, half = lane >> 4;
  int qend = qbase + 15; if (qend > Sc - 1) qend = Sc - 1;
  int nkc = qend / 32 + 1; // 32-key chunks (uniform across block)

  if (wave == 0) // preload K chunk 0:  tile [y=32 keys][x=256 dims]
    tdm_load_2d(Kp, lds_off(kbuf[0]), HDc, Spc, HDc, 32, HDc, 6, 3,
                kbuf[0], KLD);

  for (int kc = 0; kc < nkc; ++kc) {
    int kb = kc * 32, cur = kc & 1;
    if (wave == 0) tdm_wait<0>();      // K(kc) resident in kbuf[cur]
    __syncthreads();                   // publish; prior vbuf readers done
    if (wave == 0) {
      // V(kc): tile [y=256 dims][x=32 keys] from V^T (row stride Spc)
      tdm_load_2d(Vp + kb, lds_off(vbuf), (unsigned)(Spc - kb), HDc, 32, HDc,
                  Spc, 3, 3, vbuf, VLD);
      if (kc + 1 < nkc)                // prefetch K(kc+1)
        tdm_load_2d(Kp + (size_t)(kb + 32) * HDc, lds_off(kbuf[cur ^ 1]),
                    HDc, (unsigned)(Spc - (kb + 32)), HDc, 32, HDc, 6, 3,
                    kbuf[cur ^ 1], KLD);
    }
    // ---- scores: S[16,32] = Q[16,256] @ K^T from LDS ----
    const bf16_t* kt = kbuf[cur];
    v8f s0 = vzero8(), s1 = vzero8();
    for (int d = 0; d < 8; ++d) {
      v16bf a  = load_a16(Qp + d * 32, HDc, lane);
      v16bf b0 = load_b16(kt + d * 32, KLD, lane);
      v16bf b1 = load_b16(kt + 16 * KLD + d * 32, KLD, lane);
      s0 = wmma_bf16(a, b0, s0);
      s1 = wmma_bf16(a, b1, s1);
    }
    // ---- causal mask + online softmax (fp32) ----
    int kg0 = kb + n, kg1 = kb + 16 + n;
    float p0[8], p1[8];
    for (int r = 0; r < 8; ++r) {
      int qg = qbase + r + 8 * half;
      float v0 = (kg0 <= qg) ? s0[r] : -INFINITY; // scale folded into Q
      float v1 = (kg1 <= qg) ? s1[r] : -INFINITY;
      float cm = fmaxf(v0, v1);
      for (int off = 1; off < 16; off <<= 1)      // stays within 16-lane half
        cm = fmaxf(cm, __shfl_xor(cm, off));
      float mn = fmaxf(mi[r], cm);
      float al = __expf(mi[r] - mn);
      p0[r] = __expf(v0 - mn);
      p1[r] = __expf(v1 - mn);
      float rs = p0[r] + p1[r];
      for (int off = 1; off < 16; off <<= 1)
        rs += __shfl_xor(rs, off);
      li[r] = li[r] * al + rs;
      mi[r] = mn;
      for (int dt = 0; dt < 8; ++dt) acc[dt][r] *= al;
    }
    // ---- P: C-layout fp32 -> A-layout bf16 via LDS ----
    bf16_t* pw = plds[wave];
    for (int r = 0; r < 8; ++r) {
      int m = r + 8 * half;
      pw[m * 40 + n]      = (bf16_t)p0[r];
      pw[m * 40 + 16 + n] = (bf16_t)p1[r];
    }
    __syncthreads();
    v16bf pa = load_a16(pw, 40, lane);
    if (wave == 0) {                   // V(kc) done (TDM completes in order)
      if (kc + 1 < nkc) tdm_wait<1>(); else tdm_wait<0>();
    }
    __syncthreads();                   // publish vbuf
    // ---- O[:, wave half] += P[16,32] @ V[32,128] from LDS ----
    const bf16_t* vt = vbuf + (size_t)(wave * 128) * VLD;
    for (int dt = 0; dt < 8; ++dt) {
      v16bf bv = load_b16(vt + (size_t)(dt * 16) * VLD, VLD, lane);
      acc[dt] = wmma_bf16(pa, bv, acc[dt]);
    }
  }

  // ---- finalize: O = acc / l, merge heads into [B*S, 512] bf16 ----
  int bb = bh >> 1, hh = bh & 1;
  for (int r = 0; r < 8; ++r) {
    int qg = qbase + r + 8 * half;
    if (qg >= Sc) continue;
    float inv = 1.0f / li[r];
    size_t rowo = ((size_t)(bb * Sc + qg)) * Dc + hh * HDc + wave * 128 + n;
    for (int dt = 0; dt < 8; ++dt)
      O[rowo + dt * 16] = (bf16_t)(acc[dt][r] * inv);
  }
}

// ---------------------------------------------------------------------------
// Output projection: Y = O @ W_out + b_out (fp32 out). 48x64 tile per wave:
// grid (398, 8), block 32.
// ---------------------------------------------------------------------------
__global__ void __launch_bounds__(32, 1)
out_gemm(const bf16_t* __restrict__ O,   // [B*S, 512] bf16
         const bf16_t* __restrict__ WT,  // [512, 512] bf16 (W^T)
         const float*  __restrict__ bias,
         float* __restrict__ Y) {
  int rt = blockIdx.x, ct = blockIdx.y;
  int lane = threadIdx.x & 31;
  const bf16_t* Ot = O  + (size_t)rt * 48 * Dc;
  const bf16_t* Wt = WT + (size_t)ct * 64 * Dc;
  v8f acc[3][4];
  for (int i = 0; i < 3; ++i)
    for (int j = 0; j < 4; ++j) acc[i][j] = vzero8();
  for (int kc = 0; kc < Dc / 32; ++kc) {
    v16bf a0 = load_a16(Ot + 0 * 16 * Dc + kc * 32, Dc, lane);
    v16bf a1 = load_a16(Ot + 1 * 16 * Dc + kc * 32, Dc, lane);
    v16bf a2 = load_a16(Ot + 2 * 16 * Dc + kc * 32, Dc, lane);
    for (int j = 0; j < 4; ++j) {
      v16bf b = load_b16(Wt + (size_t)j * 16 * Dc + kc * 32, Dc, lane);
      acc[0][j] = wmma_bf16(a0, b, acc[0][j]);
      acc[1][j] = wmma_bf16(a1, b, acc[1][j]);
      acc[2][j] = wmma_bf16(a2, b, acc[2][j]);
    }
  }
  int n = lane & 15, half = lane >> 4;
  for (int i = 0; i < 3; ++i) {
    for (int j = 0; j < 4; ++j) {
      int c = ct * 64 + j * 16 + n;
      float bv = bias[c];
      for (int r = 0; r < 8; ++r) {
        size_t g = (size_t)rt * 48 + i * 16 + r + 8 * half;
        Y[g * Dc + c] = acc[i][j][r] + bv;
      }
    }
  }
}

// ---------------------------------------------------------------------------
extern "C" void kernel_launch(void* const* d_in, const int* in_sizes, int n_in,
                              void* d_out, int out_size, void* d_ws, size_t ws_size,
                              hipStream_t stream) {
  const float* x  = (const float*)d_in[0]; // [B,S,D]
  const float* Wq = (const float*)d_in[1]; // [D, 3D]
  const float* bq = (const float*)d_in[2]; // [3D]
  const float* Wo = (const float*)d_in[3]; // [D, D]
  const float* bo = (const float*)d_in[4]; // [D]
  float* y = (float*)d_out;
  char* ws = (char*)d_ws;
  (void)in_sizes; (void)n_in; (void)out_size; (void)ws_size;

  const size_t nX   = (size_t)Bc * Sc * Dc;        // 9,781,248
  const size_t nWq  = (size_t)Dc * 3 * Dc;         // 786,432
  const size_t nWo  = (size_t)Dc * Dc;             // 262,144
  const size_t nQKV = (size_t)Bc * Hc * Spc * HDc; // 9,961,472

  size_t off = 0;
  auto take = [&](size_t elems) {
    size_t o = off;
    off += ((elems * sizeof(bf16_t)) + 255) & ~(size_t)255;
    return o;
  };
  bf16_t* Xbf = (bf16_t*)(ws + take(nX));
  bf16_t* WqT = (bf16_t*)(ws + take(nWq));
  bf16_t* WoT = (bf16_t*)(ws + take(nWo));
  bf16_t* Qb  = (bf16_t*)(ws + take(nQKV));
  bf16_t* Kb  = (bf16_t*)(ws + take(nQKV));
  bf16_t* Vt  = (bf16_t*)(ws + take(nQKV));
  bf16_t* Ob  = (bf16_t*)(ws + take(nX));

  // 1) conversions
  cvt_f32_to_bf16<<<dim3((unsigned)((nX + 255) / 256)), 256, 0, stream>>>(x, Xbf, (int)nX);
  transpose_cvt<<<dim3((unsigned)((nWq + 255) / 256)), 256, 0, stream>>>(Wq, WqT, Dc, 3 * Dc);
  transpose_cvt<<<dim3((unsigned)((nWo + 255) / 256)), 256, 0, stream>>>(Wo, WoT, Dc, Dc);

  // 2) QKV projection (bias fused, Q pre-scaled, V stored transposed)
  qkv_gemm<<<dim3(Bc * Sc / 48, 3 * Dc / 64), 32, 0, stream>>>(Xbf, WqT, bq, Qb, Kb, Vt);

  // 3) causal flash attention (TDM-staged K/V)
  flash_attn<<<dim3((Sc + 15) / 16, Bc * Hc), 64, 0, stream>>>(Qb, Kb, Vt, Ob);

  // 4) output projection
  out_gemm<<<dim3(Bc * Sc / 48, Dc / 64), 32, 0, stream>>>(Ob, WoT, bo, y);
}
